// SparseAttention_88304527606390
// MI455X (gfx1250) — compile-verified, matrix operand order checked by reading, not run on hardware
//
#include <hip/hip_runtime.h>
#include <hip/hip_bf16.h>

// ---------------------------------------------------------------------------
// Sparse attention, MI455X (gfx1250, wave32, WMMA).
//   B=2, S=4096, D=512. ~86 GFLOP vs ~170MB mandatory traffic -> compute
//   bound; use v_wmma_f32_16x16x32_f16 (f16 in, f32 acc) for every GEMM.
//   Q/K/V f16 (24MB) are L2-resident (192MB L2); K/V B-fragments stream from
//   L2 in the flash loop. Weight panels are staged to LDS with async-to-LDS
//   DMA (ASYNCcnt) double-buffering; fragments are register-double-buffered
//   so WMMAs overlap loads instead of s_wait_loadcnt 0 per op.
// ---------------------------------------------------------------------------

typedef __attribute__((ext_vector_type(16))) _Float16 v16h;
typedef __attribute__((ext_vector_type(8)))  _Float16 v8h;
typedef __attribute__((ext_vector_type(4)))  _Float16 v4h;
typedef __attribute__((ext_vector_type(8)))  float    v8f;

#define S_LEN 4096
#define D_DIM 512

#if defined(__AMDGCN__) &&                                              \
    __has_builtin(__builtin_amdgcn_global_load_async_to_lds_b128) &&    \
    __has_builtin(__builtin_amdgcn_s_wait_asynccnt)
#define HAVE_ASYNC 1
// Signature (from clang diagnostic): (v4i AS1*, v4i AS3*, imm, imm)
typedef int v4i_t __attribute__((vector_size(16)));
typedef __attribute__((address_space(1))) v4i_t* gv4i_p;
typedef __attribute__((address_space(3))) v4i_t* lv4i_p;
#else
#define HAVE_ASYNC 0
#endif

__device__ __forceinline__ void async_cp16(const _Float16* src, _Float16* lds) {
#if HAVE_ASYNC
  __builtin_amdgcn_global_load_async_to_lds_b128((gv4i_p)src, (lv4i_p)lds, 0, 0);
#else
  *(v8h*)lds = *(const v8h*)src;
#endif
}
__device__ __forceinline__ void async_wait0() {
#if HAVE_ASYNC
  __builtin_amdgcn_s_wait_asynccnt(0);
#endif
}

__device__ __forceinline__ v8f wmma_f16(v16h a, v16h b, v8f c) {
  // (neg_a, A, neg_b, B, c_mod, C, reuse_a, reuse_b)
  return __builtin_amdgcn_wmma_f32_16x16x32_f16(false, a, false, b,
                                                (short)0, c, false, false);
}

// A-fragment (16x32 f16, M=lane&15, K-halves split by lane group) or,
// identically, B-fragment (32x16 f16, N=lane&15) from a row-major source
// whose rows are contiguous in K. Two 16B chunks per lane.
__device__ __forceinline__ v16h load_frag(const _Float16* p, int ld) {
  const int lane = threadIdx.x & 31;
  const int row  = lane & 15;
  const int ko   = (lane >> 4) << 3;     // 0 or 8
  const _Float16* q = p + (size_t)row * ld + ko;
  v8h lo = *(const v8h*)(q);
  v8h hi = *(const v8h*)(q + 16);
  return __builtin_shufflevector(lo, hi, 0,1,2,3,4,5,6,7,8,9,10,11,12,13,14,15);
}

// ---------------------------------------------------------------------------
// f32 -> f16 conversion (x and the four weight matrices)
// ---------------------------------------------------------------------------
__global__ void cvt_f16_kernel(const float* __restrict__ src,
                               _Float16* __restrict__ dst, int n4) {
  int i = blockIdx.x * blockDim.x + threadIdx.x;
  if (i < n4) {
    float4 v = ((const float4*)src)[i];
    v4h o = {(_Float16)v.x, (_Float16)v.y, (_Float16)v.z, (_Float16)v.w};
    ((v4h*)dst)[i] = o;
  }
}

// ---------------------------------------------------------------------------
// WMMA GEMM: out[m][n] = sum_k A[m][k] * W[n][k] + bias[n]
//   A: [M x 512] f16 rm, W: [512 x 512] f16 rm (torch Linear layout).
//   Block: 8 waves; wave w -> 16x64 tile; block tile 128(M) x 64(N).
//   W k-panel (64 x 32) is shared by all 8 waves -> staged to LDS with
//   async-to-LDS DMA, double buffered; A fragments double buffered in regs.
//   mode 0: f16 out rm [M x 512]   (Q, K)
//   mode 1: f16 out transposed     (V -> Vt[d][m])
//   mode 2: f32 out rm [M x 512]   (final Wo projection into d_out)
// ---------------------------------------------------------------------------
#define WLD 48   // LDS row stride (halves): 96B, 16B-aligned, bank-spread

__global__ __launch_bounds__(256) void gemm_wmma_kernel(
    const _Float16* __restrict__ A, const _Float16* __restrict__ W,
    const float* __restrict__ bias, void* __restrict__ out, int M, int mode) {
  __shared__ __align__(16) _Float16 Ws[2][64][WLD];
  __shared__ __align__(16) _Float16 Ts[8][64][16];
  const int w    = threadIdx.x >> 5;
  const int lane = threadIdx.x & 31;
  const int g    = lane >> 4;
  const int n    = lane & 15;
  const int m0   = blockIdx.x * 128 + w * 16;
  const int n0   = blockIdx.y * 64;

  // cooperative W-panel staging: 64 rows x 32 halves, 16B per thread
  const int srow = threadIdx.x >> 2;
  const int scol = (threadIdx.x & 3) * 8;
  const _Float16* wpan = W + (size_t)(n0 + srow) * D_DIM + scol;

  v8f acc[4] = {};
  const _Float16* abase = A + (size_t)m0 * D_DIM;

  async_cp16(wpan, &Ws[0][srow][scol]);          // stage k=0 panel
  async_wait0();
  __syncthreads();

  v16h a_c = load_frag(abase, D_DIM);
  int buf = 0;
  for (int k = 0; k < D_DIM; k += 32) {
    if (k + 32 < D_DIM)                          // DMA next panel during WMMAs
      async_cp16(wpan + (k + 32), &Ws[buf ^ 1][srow][scol]);
    v16h a_n = a_c;
    if (k + 32 < D_DIM) a_n = load_frag(abase + k + 32, D_DIM);
    #pragma unroll
    for (int nf = 0; nf < 4; nf++) {
      v16h bf = load_frag(&Ws[buf][nf * 16][0], WLD);
      acc[nf] = wmma_f16(a_c, bf, acc[nf]);
    }
    a_c = a_n;
    async_wait0();
    __syncthreads();
    buf ^= 1;
  }

  if (mode == 0) {                               // f16 row-major
    _Float16* o = (_Float16*)out;
    #pragma unroll
    for (int nf = 0; nf < 4; nf++) {
      int col = n0 + nf * 16 + n;
      float bv = bias[col];
      #pragma unroll
      for (int i = 0; i < 8; i++)
        o[(size_t)(m0 + g * 8 + i) * D_DIM + col] = (_Float16)(acc[nf][i] + bv);
    }
  } else if (mode == 1) {                        // f16 transposed [N x M]
    #pragma unroll
    for (int nf = 0; nf < 4; nf++) {
      float bv = bias[n0 + nf * 16 + n];
      #pragma unroll
      for (int i = 0; i < 8; i++)
        Ts[w][nf * 16 + n][g * 8 + i] = (_Float16)(acc[nf][i] + bv);
    }
    __syncthreads();
    _Float16* o = (_Float16*)out + (size_t)n0 * M + m0;
    for (int r = lane; r < 64; r += 32) {
      uint4* dst = (uint4*)(o + (size_t)r * M);
      const uint4* src = (const uint4*)&Ts[w][r][0];
      dst[0] = src[0];
      dst[1] = src[1];
    }
  } else {                                       // f32 row-major (d_out)
    float* o = (float*)out;
    #pragma unroll
    for (int nf = 0; nf < 4; nf++) {
      int col = n0 + nf * 16 + n;
      float bv = bias[col];
      #pragma unroll
      for (int i = 0; i < 8; i++)
        o[(size_t)(m0 + g * 8 + i) * D_DIM + col] = acc[nf][i] + bv;
    }
  }
}

// ---------------------------------------------------------------------------
// Flash-style sparse attention.
//   Grid: (S/16, B); block: 256 threads = 8 waves; one 16-query tile per WG.
//   Per 128-key block: wave w computes score tile cols [kb+16w,+16) with
//   register-pipelined Q(LDS)/K(L2) fragments, masks+scales, shares scores
//   and row-max via LDS; WG-shared online softmax; each wave accumulates P*V
//   into its private 64-wide d-slice with V fragments preloaded one kf ahead.
//   Output stored transposed AOt[d][s] (== reference transpose+reshape,
//   flat-viewed as [S x D] for the final GEMM).
// ---------------------------------------------------------------------------
__global__ __launch_bounds__(256) void attn_kernel(
    const _Float16* __restrict__ Qh, const _Float16* __restrict__ Kh,
    const _Float16* __restrict__ Vt, const float* __restrict__ mask,
    _Float16* __restrict__ AOt) {
  __shared__ __align__(16) _Float16 Qs[16][528];   // padded: banks spread
  __shared__ float Ss[16][132];                    // masked, scaled scores
  __shared__ float tmax[8][16];
  __shared__ float mrun[16], lrun[16], mnew[16], corr[16];
  __shared__ __align__(16) _Float16 Ts[8][64][16]; // per-wave O transpose

  const int q0   = blockIdx.x * 16;
  const int b    = blockIdx.y;
  const int w    = threadIdx.x >> 5;
  const int lane = threadIdx.x & 31;
  const int g    = lane >> 4;
  const int n    = lane & 15;
  const float scale = 0.044194173824159216f;       // 1/sqrt(512)
  const float NEG   = -__builtin_inff();

  // Stage Q tile (16 x 512 f16) into LDS via async DMA, 64B/thread.
  {
    int row = threadIdx.x >> 4;
    int c   = (threadIdx.x & 15) * 32;
    const _Float16* src = Qh + ((size_t)b * S_LEN + q0 + row) * D_DIM + c;
    #pragma unroll
    for (int j = 0; j < 32; j += 8)
      async_cp16(src + j, &Qs[row][c + j]);
  }
  if (threadIdx.x < 16) { mrun[threadIdx.x] = NEG; lrun[threadIdx.x] = 0.f; }
  async_wait0();

  v8f acc[4] = {};

  for (int kb = 0; kb < S_LEN; kb += 128) {
    __syncthreads();
    // ---- Phase A: scores for key sub-tile [kb+16w, +16), pipelined --------
    v8f s = {};
    const _Float16* kbase = Kh + ((size_t)b * S_LEN + kb + 16 * w) * D_DIM;
    v16h qa = load_frag(&Qs[0][0], 528);
    v16h kf_ = load_frag(kbase, D_DIM);
    #pragma unroll
    for (int dk = 0; dk < D_DIM; dk += 32) {
      v16h qa_n = qa, kf_n = kf_;
      if (dk + 32 < D_DIM) {
        qa_n = load_frag(&Qs[0][dk + 32], 528);
        kf_n = load_frag(kbase + dk + 32, D_DIM);
      }
      s = wmma_f16(qa, kf_, s);
      qa = qa_n; kf_ = kf_n;
    }
    const float* mrow0 =
        mask + ((size_t)b * S_LEN + q0 + g * 8) * S_LEN + kb + 16 * w + n;
    __builtin_prefetch(mrow0 + 128, 0, 1);         // next key block's mask
    #pragma unroll
    for (int i = 0; i < 8; i++) {
      float mv = mrow0[(size_t)i * S_LEN];
      float sv = (mv > 0.95f) ? s[i] * scale : NEG;
      Ss[g * 8 + i][16 * w + n] = sv;
      float r = sv;                                // row-max over 16 lanes
      r = fmaxf(r, __shfl_xor(r, 1));
      r = fmaxf(r, __shfl_xor(r, 2));
      r = fmaxf(r, __shfl_xor(r, 4));
      r = fmaxf(r, __shfl_xor(r, 8));
      if (n == i) tmax[w][g * 8 + i] = r;
    }
    __syncthreads();
    // ---- Phase B: workgroup-shared online-softmax stats -------------------
    if (threadIdx.x < 16) {
      int r = threadIdx.x;
      float mx = tmax[0][r];
      #pragma unroll
      for (int t = 1; t < 8; t++) mx = fmaxf(mx, tmax[t][r]);
      float mold = mrun[r];
      float mn   = fmaxf(mold, mx);
      float c    = (mn <= -1e30f) ? 1.f : __expf(mold - mn);
      mnew[r] = mn; corr[r] = c; mrun[r] = mn;
    }
    __syncthreads();
    // ---- Phase C: rescale acc, build P A-frags, P*V (V preloaded) ---------
    #pragma unroll
    for (int nf = 0; nf < 4; nf++)
      #pragma unroll
      for (int i = 0; i < 8; i++) acc[nf][i] *= corr[g * 8 + i];

    const int row = n;                             // A-frag row = lane&15
    const int ko  = g * 8;
    const float mrow = mnew[row];
    const _Float16* vbase =
        Vt + (size_t)(64 * w) * (2 * S_LEN) + (size_t)b * S_LEN + kb;
    v16h vb_c[4];
    #pragma unroll
    for (int nf = 0; nf < 4; nf++)
      vb_c[nf] = load_frag(vbase + (size_t)(16 * nf) * (2 * S_LEN), 2 * S_LEN);

    float lsum = 0.f;
    #pragma unroll
    for (int kf = 0; kf < 4; kf++) {
      v16h p = {};
      #pragma unroll
      for (int j = 0; j < 8; j++) {
        float s0 = Ss[row][kf * 32 + ko + j];
        float s1 = Ss[row][kf * 32 + ko + 16 + j];
        float e0 = (s0 <= -1e30f) ? 0.f : __expf(s0 - mrow);
        float e1 = (s1 <= -1e30f) ? 0.f : __expf(s1 - mrow);
        lsum += e0 + e1;
        p[j]     = (_Float16)e0;
        p[8 + j] = (_Float16)e1;
      }
      v16h vb_n[4];
      if (kf < 3) {                                // preload next kf's V frags
        #pragma unroll
        for (int nf = 0; nf < 4; nf++)
          vb_n[nf] = load_frag(vbase + (kf + 1) * 32 +
                               (size_t)(16 * nf) * (2 * S_LEN), 2 * S_LEN);
      }
      #pragma unroll
      for (int nf = 0; nf < 4; nf++)
        acc[nf] = wmma_f16(p, vb_c[nf], acc[nf]);
      if (kf < 3) {
        #pragma unroll
        for (int nf = 0; nf < 4; nf++) vb_c[nf] = vb_n[nf];
      }
    }
    float rowtot = lsum + __shfl_xor(lsum, 16);
    if (w == 0 && lane < 16) lrun[lane] = lrun[lane] * corr[lane] + rowtot;
  }
  __syncthreads();

  // ---- Finalize: 1/l, per-wave LDS transpose, coalesced store to AOt[d][s].
  #pragma unroll
  for (int nf = 0; nf < 4; nf++) {
    #pragma unroll
    for (int i = 0; i < 8; i++) {
      int m = g * 8 + i;
      float l   = lrun[m];
      float inv = (l > 0.f) ? 1.f / l : 0.f;
      Ts[w][nf * 16 + n][m] = (_Float16)(acc[nf][i] * inv);
    }
  }
  __syncthreads();
  _Float16* obase = AOt + ((size_t)b * D_DIM + 64 * w) * S_LEN + q0;
  for (int r = lane; r < 64; r += 32) {
    uint4* dst = (uint4*)(obase + (size_t)r * S_LEN);
    const uint4* src = (const uint4*)&Ts[w][r][0];
    dst[0] = src[0];
    dst[1] = src[1];
  }
}

// ---------------------------------------------------------------------------
extern "C" void kernel_launch(void* const* d_in, const int* in_sizes, int n_in,
                              void* d_out, int out_size, void* d_ws,
                              size_t ws_size, hipStream_t stream) {
  (void)in_sizes; (void)n_in; (void)out_size; (void)ws_size;
  const float* x    = (const float*)d_in[0];
  const float* mask = (const float*)d_in[1];
  const float* Wq   = (const float*)d_in[2];
  const float* bq   = (const float*)d_in[3];
  const float* Wk   = (const float*)d_in[4];
  const float* bk   = (const float*)d_in[5];
  const float* Wv   = (const float*)d_in[6];
  const float* bv   = (const float*)d_in[7];
  const float* Wo   = (const float*)d_in[8];
  const float* bo   = (const float*)d_in[9];

  const int B = 2, S = S_LEN, D = D_DIM;
  const int M = B * S;                             // 8192
  const size_t MB = 1ull << 20;
  char* ws = (char*)d_ws;                          // 34 MB used
  _Float16* Qh  = (_Float16*)(ws + 0 * MB);        // 8MB [b][s][d]
  _Float16* Kh  = (_Float16*)(ws + 8 * MB);        // 8MB [b][s][d]
  _Float16* Vt  = (_Float16*)(ws + 16 * MB);       // 8MB [d][b*S+s]
  _Float16* xh  = (_Float16*)(ws + 24 * MB);       // 8MB, reused as AOt
  _Float16* AOt = xh;                              // [b][d][s] after attn
  _Float16* Wqh = (_Float16*)(ws + 32 * MB);
  _Float16* Wkh = (_Float16*)(ws + 32 * MB + 512 * 1024);
  _Float16* Wvh = (_Float16*)(ws + 33 * MB);
  _Float16* Woh = (_Float16*)(ws + 33 * MB + 512 * 1024);

  int nx4 = M * D / 4;
  cvt_f16_kernel<<<dim3((nx4 + 255) / 256), dim3(256), 0, stream>>>(x, xh, nx4);
  int nw4 = D * D / 4;
  cvt_f16_kernel<<<dim3((nw4 + 255) / 256), dim3(256), 0, stream>>>(Wq, Wqh, nw4);
  cvt_f16_kernel<<<dim3((nw4 + 255) / 256), dim3(256), 0, stream>>>(Wk, Wkh, nw4);
  cvt_f16_kernel<<<dim3((nw4 + 255) / 256), dim3(256), 0, stream>>>(Wv, Wvh, nw4);
  cvt_f16_kernel<<<dim3((nw4 + 255) / 256), dim3(256), 0, stream>>>(Wo, Woh, nw4);

  dim3 gg(M / 128, D / 64);
  gemm_wmma_kernel<<<gg, dim3(256), 0, stream>>>(xh, Wqh, bq, Qh, M, 0);
  gemm_wmma_kernel<<<gg, dim3(256), 0, stream>>>(xh, Wkh, bk, Kh, M, 0);
  gemm_wmma_kernel<<<gg, dim3(256), 0, stream>>>(xh, Wvh, bv, Vt, M, 1);

  attn_kernel<<<dim3(S / 16, B), dim3(256), 0, stream>>>(Qh, Kh, Vt, mask, AOt);

  gemm_wmma_kernel<<<gg, dim3(256), 0, stream>>>(AOt, Woh, bo, d_out, M, 2);
}
